// TransformerDecoder_7267084664854
// MI455X (gfx1250) — matrix operand order repaired and checked by reading, not compiled, not run on hardware
//
#include <hip/hip_runtime.h>
#include <hip/hip_bf16.h>

typedef __attribute__((ext_vector_type(16))) _Float16 v16h;
typedef __attribute__((ext_vector_type(4)))  _Float16 v4h;
typedef __attribute__((ext_vector_type(8)))  float    v8f;
typedef __attribute__((ext_vector_type(4)))  float    v4f;

#define B_   4
#define S_   512
#define SR_  512
#define H_   1024
#define NH_  16
#define HD_  64
#define L_   6
#define V_   32000
#define FF_  4096
#define PAD_ 2

// ---------------------------------------------------------------------------
// Embedding gather: x[row,:] = emb[tok[row],:]
// ---------------------------------------------------------------------------
__global__ void embed_kernel(const float* __restrict__ emb,
                             const int* __restrict__ tok,
                             float* __restrict__ x) {
  const long row = blockIdx.x;
  const int  t   = tok[row];
  const float* e = emb + (long)t * H_;
  float*       o = x   + row * H_;
  for (int c = threadIdx.x; c < H_; c += blockDim.x) o[c] = e[c];
}

// ---------------------------------------------------------------------------
// Generic batched GEMM via V_WMMA_F32_16X16X32_F16.
//   C[b,h] = scale * (A[b,h] @ op(B[b,h]) + bias), optional relu.
// Block = 4 waves; block tile = 32(M) x 64(N); K consumed 32/step.
// f32 operands are converted to f16 while staging into LDS in *fragment
// order*: each lane's 16 A/B values live contiguously as one v16h slot, so
// a fragment load is two ds_load_b128 (instead of 16 ds_load_u16).
// Global staging uses float4 (global_load_b128) everywhere; A and trans-B
// paths write packed ds_store_b64.
// Layouts use contiguous k order (k = 16*h + slot) for both A(16x32) and B(32x16):
//   A: lane = (m%16) + 16*(k>>4), slot = k&15
//   B: lane = (n%16) + 16*(k>>4), slot = k&15
// Requires: M%32==0, N%64==0, K%32==0 (true for all uses here).
// ---------------------------------------------------------------------------
template <int TRANSB>
__global__ __launch_bounds__(128)
void gemm_wmma_kernel(const float* __restrict__ A, const float* __restrict__ Bm,
                      const float* __restrict__ bias, float* __restrict__ C,
                      int M, int N, int K, int lda, int ldb, int ldc,
                      long sAb, long sAh, long sBb, long sBh, long sCb, long sCh,
                      int nh, float scale, int relu) {
  __shared__ v16h AsF[2][32];   // 2 M-subtiles x 32 lanes (2 KB)
  __shared__ v16h BsF[4][32];   // 4 N-subtiles x 32 lanes (4 KB)

  const int tid  = threadIdx.x;
  const int wave = tid >> 5;
  const int lane = tid & 31;
  const int lh   = lane >> 4;      // lane half
  const int lr   = lane & 15;

  const int m0   = blockIdx.y * 32;
  const int nblk = blockIdx.x * 64;
  const int bz   = blockIdx.z;
  const int b    = bz / nh;
  const int h    = bz % nh;

  const float* Ab = A  + b * sAb + h * sAh;
  const float* Bb = Bm + b * sBb + h * sBh;
  float*       Cb = C  + b * sCb + h * sCh;

  _Float16* asf = (_Float16*)AsF;
  _Float16* bsf = (_Float16*)BsF;

  v8f acc0 = {};
  v8f acc1 = {};

  for (int kk = 0; kk < K; kk += 32) {
    // ---- stage A: 32x32 f32 -> fragment-ordered f16 (2 float4 per thread)
    #pragma unroll
    for (int t = 0; t < 2; ++t) {
      int e  = tid + t * 128;              // 0..255
      int r  = e >> 3;                     // 0..31  (m-local)
      int k0 = (e & 7) * 4;                // 0,4,...,28
      v4f a = *(const v4f*)(Ab + (long)(m0 + r) * lda + (kk + k0));
      v4h hv;
      hv[0] = (_Float16)a[0]; hv[1] = (_Float16)a[1];
      hv[2] = (_Float16)a[2]; hv[3] = (_Float16)a[3];
      int sub   = r >> 4;
      int rl    = r & 15;
      int ah    = k0 >> 4;                             // A lane-half for k0
      int slot0 = k0 & 15;                             // slot = k&15
      *(v4h*)(asf + (((sub * 32) + rl + 16 * ah) * 16 + slot0)) = hv;
    }
    // ---- stage B: 32x64 f32 -> fragment-ordered f16 (4 float4 per thread)
    if (TRANSB) {
      #pragma unroll
      for (int t = 0; t < 4; ++t) {
        int e  = tid + t * 128;            // 0..511
        int c  = e >> 3;                   // 0..63  (n-local)
        int k0 = (e & 7) * 4;              // 0,4,...,28
        v4f bb = *(const v4f*)(Bb + (long)(nblk + c) * ldb + (kk + k0));
        v4h hv;
        hv[0] = (_Float16)bb[0]; hv[1] = (_Float16)bb[1];
        hv[2] = (_Float16)bb[2]; hv[3] = (_Float16)bb[3];
        int w = c >> 4, cl = c & 15;
        int bh = k0 >> 4, slot0 = k0 & 15;
        *(v4h*)(bsf + (((w * 32) + cl + 16 * bh) * 16 + slot0)) = hv;
      }
    } else {
      #pragma unroll
      for (int t = 0; t < 4; ++t) {
        int e  = tid + t * 128;            // 0..511
        int r  = e >> 4;                   // 0..31  (k-local)
        int c0 = (e & 15) * 4;             // 0,4,...,60
        v4f bb = *(const v4f*)(Bb + (long)(kk + r) * ldb + (nblk + c0));
        int bh = r >> 4, slot = r & 15;
        int w = c0 >> 4, cl0 = c0 & 15;
        #pragma unroll
        for (int j = 0; j < 4; ++j)
          bsf[((w * 32) + (cl0 + j) + 16 * bh) * 16 + slot] = (_Float16)bb[j];
      }
    }
    __syncthreads();

    v16h bf = BsF[wave][lane];   // 2x ds_load_b128
    v16h a0 = AsF[0][lane];
    v16h a1 = AsF[1][lane];
    acc0 = __builtin_amdgcn_wmma_f32_16x16x32_f16(false, a0, false, bf,
                                                  (short)0, acc0, false, false);
    acc1 = __builtin_amdgcn_wmma_f32_16x16x32_f16(false, a1, false, bf,
                                                  (short)0, acc1, false, false);
    __syncthreads();
  }

  // C/D layout: VGPR r -> M = r + lh*8, N = lr (per 16x16 subtile).
  const int   col = nblk + wave * 16 + lr;
  const float bv  = bias ? bias[col] : 0.0f;
  #pragma unroll
  for (int r = 0; r < 8; ++r) {
    int   row = m0 + r + lh * 8;
    float v0  = (acc0[r] + bv) * scale;
    float v1  = (acc1[r] + bv) * scale;
    if (relu) { v0 = fmaxf(v0, 0.0f); v1 = fmaxf(v1, 0.0f); }
    Cb[(long)row * ldc + col]        = v0;
    Cb[(long)(row + 16) * ldc + col] = v1;
  }
}

// ---------------------------------------------------------------------------
// Attention softmax (in-place on att rows), fusing:
//  - relative-position skew: att[i,j] += qp[i, j-i+S-1]   (self-attn only)
//  - causal mask (j>i -> -inf)
//  - key-padding mask from token ids (tok[b,j]==PAD -> -inf)
// One block (256 thr) per (b,h,i) row of `ncols` (=512 here).
// ---------------------------------------------------------------------------
__global__ void attn_softmax_kernel(float* __restrict__ att,
                                    const float* __restrict__ qp,
                                    const int* __restrict__ tok,
                                    int ncols, int causal) {
  const int i = blockIdx.x, h = blockIdx.y, b = blockIdx.z;
  float* row = att + (((long)b * NH_ + h) * S_ + i) * (long)ncols;
  const float* qpr = qp ? qp + (((long)b * NH_ + h) * S_ + i) * (long)S_ : nullptr;
  const int tid = threadIdx.x;
  __shared__ float red[256];
  const float NEG = -__builtin_inff();

  float vals[2];                         // ncols == 512 -> 2 per thread
  const int nit = ncols / 256;
  float mx = NEG;
  for (int t = 0; t < nit; ++t) {
    int   j = tid + t * 256;
    float v = row[j];
    if (qpr && j <= i) v += qpr[j - i + (S_ - 1)];
    if (causal && j > i) v = NEG;
    if (tok[b * ncols + j] == PAD_) v = NEG;
    vals[t] = v;
    mx = fmaxf(mx, v);
  }
  red[tid] = mx; __syncthreads();
  for (int s = 128; s > 0; s >>= 1) {
    if (tid < s) red[tid] = fmaxf(red[tid], red[tid + s]);
    __syncthreads();
  }
  mx = red[0]; __syncthreads();

  float sum = 0.0f;
  for (int t = 0; t < nit; ++t) { vals[t] = __expf(vals[t] - mx); sum += vals[t]; }
  red[tid] = sum; __syncthreads();
  for (int s = 128; s > 0; s >>= 1) {
    if (tid < s) red[tid] += red[tid + s];
    __syncthreads();
  }
  const float inv = 1.0f / red[0];
  for (int t = 0; t < nit; ++t) row[tid + t * 256] = vals[t] * inv;
}

// ---------------------------------------------------------------------------
// out = LayerNorm(a + resid) * g + beta          (one block per row of H)
// ---------------------------------------------------------------------------
__global__ void add_layernorm_kernel(const float* __restrict__ a,
                                     const float* __restrict__ resid,
                                     const float* __restrict__ g,
                                     const float* __restrict__ beta,
                                     float* __restrict__ out) {
  const long row = blockIdx.x;
  const int  tid = threadIdx.x;
  const float* ar = a     + row * H_;
  const float* rr = resid + row * H_;
  float*       orow = out + row * H_;
  __shared__ float red[256];

  float vals[H_ / 256];                  // 4
  float s = 0.0f;
  #pragma unroll
  for (int t = 0; t < H_ / 256; ++t) {
    int c   = tid + t * 256;
    vals[t] = ar[c] + rr[c];
    s += vals[t];
  }
  red[tid] = s; __syncthreads();
  for (int k = 128; k > 0; k >>= 1) { if (tid < k) red[tid] += red[tid + k]; __syncthreads(); }
  const float mean = red[0] * (1.0f / H_); __syncthreads();

  float sq = 0.0f;
  #pragma unroll
  for (int t = 0; t < H_ / 256; ++t) { float d = vals[t] - mean; sq += d * d; }
  red[tid] = sq; __syncthreads();
  for (int k = 128; k > 0; k >>= 1) { if (tid < k) red[tid] += red[tid + k]; __syncthreads(); }
  const float rstd = rsqrtf(red[0] * (1.0f / H_) + 1e-5f);

  #pragma unroll
  for (int t = 0; t < H_ / 256; ++t) {
    int c = tid + t * 256;
    orow[c] = (vals[t] - mean) * rstd * g[c] + beta[c];
  }
}

// ---------------------------------------------------------------------------
// In-place log_softmax over V=32000 columns (one block per row).
// ---------------------------------------------------------------------------
__global__ void log_softmax_kernel(float* __restrict__ x) {
  const long row = blockIdx.x;
  float* r = x + row * (long)V_;
  const int tid = threadIdx.x;
  __shared__ float red[256];

  float mx = -3.4e38f;
  for (int c = tid; c < V_; c += 256) mx = fmaxf(mx, r[c]);
  red[tid] = mx; __syncthreads();
  for (int k = 128; k > 0; k >>= 1) { if (tid < k) red[tid] = fmaxf(red[tid], red[tid + k]); __syncthreads(); }
  mx = red[0]; __syncthreads();

  float s = 0.0f;
  for (int c = tid; c < V_; c += 256) s += __expf(r[c] - mx);
  red[tid] = s; __syncthreads();
  for (int k = 128; k > 0; k >>= 1) { if (tid < k) red[tid] += red[tid + k]; __syncthreads(); }
  const float lse = __logf(red[0]) + mx;

  for (int c = tid; c < V_; c += 256) r[c] = r[c] - lse;
}

// ---------------------------------------------------------------------------
extern "C" void kernel_launch(void* const* d_in, const int* in_sizes, int n_in,
                              void* d_out, int out_size, void* d_ws, size_t ws_size,
                              hipStream_t stream) {
  (void)in_sizes; (void)n_in; (void)out_size; (void)ws_size;

  const float* enc     = (const float*)d_in[0];
  const int*   dec_tok = (const int*)d_in[1];
  const int*   enc_tok = (const int*)d_in[2];
  const float* emb     = (const float*)d_in[3];
  const float* wq1 = (const float*)d_in[4],  *bq1 = (const float*)d_in[5];
  const float* wk1 = (const float*)d_in[6],  *bk1 = (const float*)d_in[7];
  const float* wv1 = (const float*)d_in[8],  *bv1 = (const float*)d_in[9];
  const float* wo1 = (const float*)d_in[10], *bo1 = (const float*)d_in[11];
  const float* pos1 = (const float*)d_in[12];
  const float* wq2 = (const float*)d_in[13], *bq2 = (const float*)d_in[14];
  const float* wk2 = (const float*)d_in[15], *bk2 = (const float*)d_in[16];
  const float* wv2 = (const float*)d_in[17], *bv2 = (const float*)d_in[18];
  const float* wo2 = (const float*)d_in[19], *bo2 = (const float*)d_in[20];
  const float* ln1g = (const float*)d_in[21], *ln1b = (const float*)d_in[22];
  const float* ln2g = (const float*)d_in[23], *ln2b = (const float*)d_in[24];
  const float* ln3g = (const float*)d_in[25], *ln3b = (const float*)d_in[26];
  const float* wf1 = (const float*)d_in[27], *bf1 = (const float*)d_in[28];
  const float* wf2 = (const float*)d_in[29], *bf2 = (const float*)d_in[30];
  const float* wout = (const float*)d_in[31], *bout = (const float*)d_in[32];

  const long BSH  = (long)B_ * S_ * H_;
  const long BSFF = (long)B_ * S_ * FF_;
  const long BNSS = (long)B_ * NH_ * S_ * S_;

  float* ws  = (float*)d_ws;
  float* x   = ws;            // [B,S,H]  layer input
  float* x1  = x  + BSH;      // after self-attn LN
  float* x2  = x1 + BSH;      // after cross-attn LN
  float* q   = x2 + BSH;
  float* k   = q  + BSH;
  float* v   = k  + BSH;
  float* ctx = v  + BSH;      // attention context (pre-Wo)
  float* prj = ctx + BSH;     // Wo / FFN2 output
  float* ff  = prj + BSH;     // [B,S,FF]
  float* att = ff + BSFF;     // [B,NH,S,S]
  float* qpb = att + BNSS;    // [B,NH,S,S]  (Q @ P^T)

  auto gemm = [&](const float* A, const float* W, const float* bias, float* C,
                  int M, int N, int K, int lda, int ldb, int ldc,
                  long sAb, long sAh, long sBb, long sBh, long sCb, long sCh,
                  int nh, int nb, int transB, float scale, int relu) {
    dim3 g(N / 64, M / 32, nb);
    if (transB)
      gemm_wmma_kernel<1><<<g, 128, 0, stream>>>(A, W, bias, C, M, N, K,
                                                 lda, ldb, ldc,
                                                 sAb, sAh, sBb, sBh, sCb, sCh,
                                                 nh, scale, relu);
    else
      gemm_wmma_kernel<0><<<g, 128, 0, stream>>>(A, W, bias, C, M, N, K,
                                                 lda, ldb, ldc,
                                                 sAb, sAh, sBb, sBh, sCb, sCh,
                                                 nh, scale, relu);
  };

  const float qscale = 0.125f;  // 1/sqrt(HD)
  const long  SH = (long)S_ * H_;
  const long  SS = (long)S_ * S_;

  embed_kernel<<<B_ * S_, 256, 0, stream>>>(emb, dec_tok, x);

  for (int i = 0; i < L_; ++i) {
    const long wo = (long)i * H_ * H_;
    const long bo = (long)i * H_;

    // ---------------- self attention ----------------
    gemm(x, wq1 + wo, bq1 + bo, q, B_*S_, H_, H_, H_, H_, H_, 0,0,0,0,0,0, 1, 1, 0, qscale, 0);
    gemm(x, wk1 + wo, bk1 + bo, k, B_*S_, H_, H_, H_, H_, H_, 0,0,0,0,0,0, 1, 1, 0, 1.0f, 0);
    gemm(x, wv1 + wo, bv1 + bo, v, B_*S_, H_, H_, H_, H_, H_, 0,0,0,0,0,0, 1, 1, 0, 1.0f, 0);
    // att[b,h] = q[b,h] @ k[b,h]^T
    gemm(q, k, nullptr, att, S_, S_, HD_, H_, H_, S_,
         SH, HD_, SH, HD_, (long)NH_*SS, SS, NH_, B_*NH_, 1, 1.0f, 0);
    // qp[b,h] = q[b,h] @ P^T   (P = pos1[i][0:S], shared across batch)
    gemm(q, pos1 + (long)i * (2*S_-1) * HD_, nullptr, qpb, S_, S_, HD_, H_, HD_, S_,
         SH, HD_, 0, 0, (long)NH_*SS, SS, NH_, B_*NH_, 1, 1.0f, 0);
    attn_softmax_kernel<<<dim3(S_, NH_, B_), 256, 0, stream>>>(att, qpb, dec_tok, S_, 1);
    // ctx[b,:,h*HD:] = att[b,h] @ v[b,h]
    gemm(att, v, nullptr, ctx, S_, HD_, S_, S_, H_, H_,
         (long)NH_*SS, SS, SH, HD_, SH, HD_, NH_, B_*NH_, 0, 1.0f, 0);
    gemm(ctx, wo1 + wo, bo1 + bo, prj, B_*S_, H_, H_, H_, H_, H_, 0,0,0,0,0,0, 1, 1, 0, 1.0f, 0);
    add_layernorm_kernel<<<B_ * S_, 256, 0, stream>>>(prj, x, ln1g + bo, ln1b + bo, x1);

    // ---------------- cross attention ----------------
    gemm(x1,  wq2 + wo, bq2 + bo, q, B_*S_,  H_, H_, H_, H_, H_, 0,0,0,0,0,0, 1, 1, 0, qscale, 0);
    gemm(enc, wk2 + wo, bk2 + bo, k, B_*SR_, H_, H_, H_, H_, H_, 0,0,0,0,0,0, 1, 1, 0, 1.0f, 0);
    gemm(enc, wv2 + wo, bv2 + bo, v, B_*SR_, H_, H_, H_, H_, H_, 0,0,0,0,0,0, 1, 1, 0, 1.0f, 0);
    gemm(q, k, nullptr, att, S_, SR_, HD_, H_, H_, SR_,
         SH, HD_, (long)SR_*H_, HD_, (long)NH_*S_*SR_, (long)S_*SR_, NH_, B_*NH_, 1, 1.0f, 0);
    attn_softmax_kernel<<<dim3(S_, NH_, B_), 256, 0, stream>>>(att, nullptr, enc_tok, SR_, 0);
    gemm(att, v, nullptr, ctx, S_, HD_, SR_, SR_, H_, H_,
         (long)NH_*S_*SR_, (long)S_*SR_, (long)SR_*H_, HD_, SH, HD_, NH_, B_*NH_, 0, 1.0f, 0);
    gemm(ctx, wo2 + wo, bo2 + bo, prj, B_*S_, H_, H_, H_, H_, H_, 0,0,0,0,0,0, 1, 1, 0, 1.0f, 0);
    add_layernorm_kernel<<<B_ * S_, 256, 0, stream>>>(prj, x1, ln2g + bo, ln2b + bo, x2);

    // ---------------- feed forward ----------------
    gemm(x2, wf1 + (long)i * H_ * FF_, bf1 + (long)i * FF_, ff,
         B_*S_, FF_, H_, H_, FF_, FF_, 0,0,0,0,0,0, 1, 1, 0, 1.0f, 1);
    gemm(ff, wf2 + (long)i * FF_ * H_, bf2 + bo, prj,
         B_*S_, H_, FF_, FF_, H_, H_, 0,0,0,0,0,0, 1, 1, 0, 1.0f, 0);
    add_layernorm_kernel<<<B_ * S_, 256, 0, stream>>>(prj, x2, ln3g + bo, ln3b + bo, x);
  }

  // ---------------- output projection + log_softmax ----------------
  float* out = (float*)d_out;
  gemm(x, wout, bout, out, B_*S_, V_, H_, H_, V_, V_, 0,0,0,0,0,0, 1, 1, 0, 1.0f, 0);
  log_softmax_kernel<<<B_ * S_, 256, 0, stream>>>(out);
}
